// ComplexityDecoderLayer_87016037417302
// MI455X (gfx1250) — matrix operand order repaired and checked by reading, not compiled, not run on hardware
//
#include <hip/hip_runtime.h>
#include <cstdint>

#define NTOK 2048
#define DMODEL 1024
#define NH 16
#define NKV 4
#define DHEAD 64
#define NEXP 8
#define FDIM 2048
#define CHID 64

typedef __bf16 bf16_t;
typedef __attribute__((ext_vector_type(16))) __bf16 v16bf;
typedef __attribute__((ext_vector_type(8)))  float  v8f;

union FragU { uint4 u[2]; v16bf v; };

__device__ __forceinline__ v16bf ld_frag(const bf16_t* p0, const bf16_t* p1) {
  FragU f;
  f.u[0] = *reinterpret_cast<const uint4*>(p0);
  f.u[1] = *reinterpret_cast<const uint4*>(p1);
  return f.v;
}

__device__ __forceinline__ v8f wmma_bf16(v16bf a, v16bf b, v8f c) {
  return __builtin_amdgcn_wmma_f32_16x16x32_bf16(false, a, false, b, (short)0, c, false, false);
}

// ---------------------------------------------------------------------------
// Generic dense GEMM: C[M,Nc] = A1@B1 (+ A2@B2), A bf16 row-major, B f32 row-
// major (converted to bf16 while staged into LDS transposed, padded stride 40
// halfs -> conflict-free ds_load_b128 fragments).
// Register blocking: each wave owns 16 rows x 64 cols (4 accumulators, A
// fragment reused across 4 B fragments); a block of 4 waves covers 64x64 and
// shares one LDS B tile -> 4x WMMA per global/LDS-stage vs naive 16x16.
// mode 0: store f32 ; 1: +bias, f32 ; 2: +bias, silu, bf16
// ---------------------------------------------------------------------------
__global__ __launch_bounds__(128) void gemm_dense(
    const bf16_t* __restrict__ A1, const float* __restrict__ B1,
    const bf16_t* __restrict__ A2, const float* __restrict__ B2,
    int Nc, int K, int lda, int ldb,
    const float* __restrict__ bias, float* __restrict__ outF,
    bf16_t* __restrict__ outH, int mode)
{
  __shared__ bf16_t Bs[2][64 * 40];
  const int tid = threadIdx.x;
  const int wid = tid >> 5, lane = tid & 31;
  const int g = lane >> 4, mloc = lane & 15;
  const int rw0 = blockIdx.y * 64 + wid * 16;   // this wave's 16 rows
  const int col0 = blockIdx.x * 64;             // block's 64 cols (shared)
  const bool dual = (A2 != nullptr);

  v8f zero = {0.f,0.f,0.f,0.f,0.f,0.f,0.f,0.f};
  v8f acc[4];
  #pragma unroll
  for (int j = 0; j < 4; j++) acc[j] = zero;

  for (int kt = 0; kt < K; kt += 32) {
    __syncthreads();
    for (int i = tid; i < 32 * 64; i += 128) {
      int kk = i >> 6, c = i & 63;
      Bs[0][c * 40 + kk] = (bf16_t)B1[(size_t)(kt + kk) * ldb + col0 + c];
      if (dual) Bs[1][c * 40 + kk] = (bf16_t)B2[(size_t)(kt + kk) * ldb + col0 + c];
      if (kt + 32 < K) __builtin_prefetch(&B1[(size_t)(kt + 32 + kk) * ldb + col0 + c], 0, 1);
    }
    __syncthreads();
    const bf16_t* ar = A1 + (size_t)(rw0 + mloc) * lda + kt;
    v16bf a = ld_frag(ar + 16 * g, ar + 16 * g + 8);
    #pragma unroll
    for (int j = 0; j < 4; j++) {
      const bf16_t* bp = &Bs[0][(j * 16 + mloc) * 40 + 16 * g];
      acc[j] = wmma_bf16(a, ld_frag(bp, bp + 8), acc[j]);
    }
    if (dual) {
      const bf16_t* ar2 = A2 + (size_t)(rw0 + mloc) * lda + kt;
      v16bf a2 = ld_frag(ar2 + 16 * g, ar2 + 16 * g + 8);
      #pragma unroll
      for (int j = 0; j < 4; j++) {
        const bf16_t* bp2 = &Bs[1][(j * 16 + mloc) * 40 + 16 * g];
        acc[j] = wmma_bf16(a2, ld_frag(bp2, bp2 + 8), acc[j]);
      }
    }
  }

  #pragma unroll
  for (int j = 0; j < 4; j++) {
    const int col = col0 + j * 16 + mloc;
    #pragma unroll
    for (int r = 0; r < 8; r++) {
      int row = rw0 + r + 8 * g;
      float v = acc[j][r];
      if (mode >= 1) v += bias[col];
      if (mode == 2) {
        float s = v / (1.f + __expf(-v));
        outH[(size_t)row * Nc + col] = (bf16_t)s;
      } else {
        outF[(size_t)row * Nc + col] = v;
      }
    }
  }
}

// ---------------------------------------------------------------------------
// RMSNorm over last dim D, output bf16
// ---------------------------------------------------------------------------
__global__ __launch_bounds__(256) void rmsnorm_kernel(
    const float* __restrict__ x, const float* __restrict__ w, bf16_t* __restrict__ out)
{
  int n = blockIdx.x, tid = threadIdx.x;
  __shared__ float red[8];
  __shared__ float sinv;
  const float* row = x + (size_t)n * DMODEL;
  float s = 0.f;
  for (int i = tid; i < DMODEL; i += 256) { float v = row[i]; s += v * v; }
  #pragma unroll
  for (int o = 16; o >= 1; o >>= 1) s += __shfl_xor(s, o, 32);
  if ((tid & 31) == 0) red[tid >> 5] = s;
  __syncthreads();
  if (tid == 0) {
    float t = 0.f;
    #pragma unroll
    for (int i = 0; i < 8; i++) t += red[i];
    sinv = rsqrtf(t / (float)DMODEL + 1e-6f);
  }
  __syncthreads();
  float iv = sinv;
  for (int i = tid; i < DMODEL; i += 256)
    out[(size_t)n * DMODEL + i] = (bf16_t)(row[i] * iv * w[i]);
}

__global__ void cvt_kernel(const float* __restrict__ x, bf16_t* __restrict__ y, int n) {
  int i = blockIdx.x * blockDim.x + threadIdx.x;
  if (i < n) y[i] = (bf16_t)x[i];
}

// v [N][KV*DH] f32 -> vT [KV*DH][N] bf16
__global__ void vtrans_kernel(const float* __restrict__ v, bf16_t* __restrict__ vT) {
  int i = blockIdx.x * blockDim.x + threadIdx.x;
  if (i >= NKV * DHEAD * NTOK) return;
  int n = i & (NTOK - 1);
  int c = i >> 11;
  vT[(size_t)c * NTOK + n] = (bf16_t)v[(size_t)n * (NKV * DHEAD) + c];
}

// ---------------------------------------------------------------------------
// Per-(token, head) RMSNorm + RoPE for q and k; q is pre-scaled by 1/sqrt(DH)
// One wave per (token, head). Lane j handles the rope pair (j, j+32).
// ---------------------------------------------------------------------------
__global__ __launch_bounds__(32) void ropeqk_kernel(
    const float* __restrict__ q, const float* __restrict__ k,
    const float* __restrict__ qw, const float* __restrict__ kw,
    const int* __restrict__ pos, bf16_t* __restrict__ qb, bf16_t* __restrict__ kb)
{
  int n = blockIdx.x, hh = blockIdx.y, lane = threadIdx.x;
  const float* src; const float* w; bf16_t* dst; float scale;
  if (hh < NH) {
    src = q + ((size_t)n * NH + hh) * DHEAD; w = qw;
    dst = qb + ((size_t)n * NH + hh) * DHEAD; scale = 0.125f;
  } else {
    int kh = hh - NH;
    src = k + ((size_t)n * NKV + kh) * DHEAD; w = kw;
    dst = kb + ((size_t)n * NKV + kh) * DHEAD; scale = 1.0f;
  }
  float x1 = src[lane], x2 = src[lane + 32];
  float ss = x1 * x1 + x2 * x2;
  #pragma unroll
  for (int o = 16; o >= 1; o >>= 1) ss += __shfl_xor(ss, o, 32);
  float inv = rsqrtf(ss / 64.0f + 1e-6f);
  float xn1 = x1 * inv * w[lane], xn2 = x2 * inv * w[lane + 32];
  float ang = (float)pos[n] * powf(10000.0f, -(float)lane / 32.0f);
  float c = cosf(ang), s = sinf(ang);
  dst[lane]      = (bf16_t)((xn1 * c - xn2 * s) * scale);
  dst[lane + 32] = (bf16_t)((xn2 * c + xn1 * s) * scale);
}

// ---------------------------------------------------------------------------
// Flash attention: one wave owns (head h, 16-row tile). Streams keys in blocks
// of 32, S via 4 WMMA, online softmax via 16-lane shfl reductions, P staged
// through per-wave LDS to form the A fragment, P@V via 4 WMMA against vT.
// ---------------------------------------------------------------------------
__global__ __launch_bounds__(128) void attn_kernel(
    const bf16_t* __restrict__ qb, const bf16_t* __restrict__ kb,
    const bf16_t* __restrict__ vT, float* __restrict__ oraw)
{
  __shared__ bf16_t Pb[4][16 * 40];
  const int wid = threadIdx.x >> 5, lane = threadIdx.x & 31;
  const int wg = blockIdx.x * 4 + wid;          // 0 .. 2047
  const int h = wg >> 7;                        // head
  const int rt = wg & 127;                      // row tile
  const int kvh = h >> 2;                       // GQA map (groups = 4)
  const int g = lane >> 4, mloc = lane & 15;
  bf16_t* P = Pb[wid];

  const bf16_t* qrow = qb + ((size_t)(rt * 16 + mloc) * NH + h) * DHEAD;
  v16bf aq0 = ld_frag(qrow + 16 * g,      qrow + 16 * g + 8);
  v16bf aq1 = ld_frag(qrow + 32 + 16 * g, qrow + 32 + 16 * g + 8);

  v8f zero = {0.f,0.f,0.f,0.f,0.f,0.f,0.f,0.f};
  v8f acc[4];
  #pragma unroll
  for (int t = 0; t < 4; t++) acc[t] = zero;
  float mrun[8], lrun[8];
  #pragma unroll
  for (int r = 0; r < 8; r++) { mrun[r] = -3.0e38f; lrun[r] = 0.f; }

  for (int kb0 = 0; kb0 < NTOK; kb0 += 32) {
    v8f s0 = zero, s1 = zero;
    {
      const bf16_t* kr = kb + ((size_t)(kb0 + mloc) * NKV + kvh) * DHEAD;
      v16bf b0 = ld_frag(kr + 16 * g, kr + 16 * g + 8);
      v16bf b1 = ld_frag(kr + 32 + 16 * g, kr + 32 + 16 * g + 8);
      s0 = wmma_bf16(aq0, b0, s0);
      s0 = wmma_bf16(aq1, b1, s0);
    }
    {
      const bf16_t* kr = kb + ((size_t)(kb0 + 16 + mloc) * NKV + kvh) * DHEAD;
      v16bf b0 = ld_frag(kr + 16 * g, kr + 16 * g + 8);
      v16bf b1 = ld_frag(kr + 32 + 16 * g, kr + 32 + 16 * g + 8);
      s1 = wmma_bf16(aq0, b0, s1);
      s1 = wmma_bf16(aq1, b1, s1);
    }
    float sf[8];
    #pragma unroll
    for (int r = 0; r < 8; r++) {
      float v0 = s0[r], v1 = s1[r];
      float mx = fmaxf(v0, v1);
      #pragma unroll
      for (int o = 1; o < 16; o <<= 1) mx = fmaxf(mx, __shfl_xor(mx, o, 32));
      float mnew = fmaxf(mrun[r], mx);
      float scl = __expf(mrun[r] - mnew);
      float p0 = __expf(v0 - mnew), p1 = __expf(v1 - mnew);
      float rs = p0 + p1;
      #pragma unroll
      for (int o = 1; o < 16; o <<= 1) rs += __shfl_xor(rs, o, 32);
      lrun[r] = lrun[r] * scl + rs;
      mrun[r] = mnew;
      sf[r] = scl;
      int mrow = r + 8 * g;
      P[mrow * 40 + mloc]      = (bf16_t)p0;
      P[mrow * 40 + 16 + mloc] = (bf16_t)p1;
    }
    #pragma unroll
    for (int t = 0; t < 4; t++)
      #pragma unroll
      for (int r = 0; r < 8; r++) acc[t][r] *= sf[r];

    v16bf pa = ld_frag(&P[mloc * 40 + 16 * g], &P[mloc * 40 + 16 * g + 8]);
    #pragma unroll
    for (int t = 0; t < 4; t++) {
      const bf16_t* vr = vT + ((size_t)(kvh * DHEAD + t * 16 + mloc)) * NTOK + kb0 + 16 * g;
      v16bf vb = ld_frag(vr, vr + 8);
      acc[t] = wmma_bf16(pa, vb, acc[t]);
    }
  }

  #pragma unroll
  for (int t = 0; t < 4; t++)
    #pragma unroll
    for (int r = 0; r < 8; r++) {
      int row = rt * 16 + r + 8 * g;
      int col = h * DHEAD + t * 16 + mloc;
      oraw[(size_t)row * (NH * DHEAD) + col] = acc[t][r] / lrun[r];
    }
}

// ---------------------------------------------------------------------------
// Dynamics / control update
// ---------------------------------------------------------------------------
__global__ void update_kernel(
    const float* __restrict__ hidden, const float* __restrict__ o,
    const float* __restrict__ vel, const float* __restrict__ co,
    const float* __restrict__ mu, float* __restrict__ out_h, float* __restrict__ out_v)
{
  int i = blockIdx.x * blockDim.x + threadIdx.x;
  if (i >= NTOK * DMODEL) return;
  int n = i >> 10, d = i & 1023;
  const float* c = co + (size_t)n * 3 * DMODEL;
  float ar = c[d], br = c[DMODEL + d], gr = c[2 * DMODEL + d];
  float alpha = 1.f / (1.f + __expf(-ar));
  float sp = br > 20.f ? br : log1pf(__expf(br));
  float beta = fminf(sp, 2.f);
  float gate = 1.f / (1.f + __expf(-gr));
  float ov = o[i], mc = mu[i];
  float vn = alpha * vel[i] - beta * (ov - mc);
  vn = fminf(fmaxf(vn, -10.f), 10.f);
  out_v[i] = vn;
  out_h[i] = hidden[i] + ov + 0.1f * gate * vn;
}

// ---------------------------------------------------------------------------
// MoE routing: logits, biased argmax, expert histogram
// meta layout: [0..7]=counts, [8..15]=cursor, [16..24]=offsets
// ---------------------------------------------------------------------------
__global__ void router_kernel(
    const float* __restrict__ mu, const float* __restrict__ wr,
    const int* __restrict__ tok, int* __restrict__ eid, int* __restrict__ meta)
{
  int n = blockIdx.x * blockDim.x + threadIdx.x;
  if (n >= NTOK) return;
  float acc[8];
  #pragma unroll
  for (int e = 0; e < 8; e++) acc[e] = 0.f;
  const float* m = mu + (size_t)n * DMODEL;
  for (int k = 0; k < DMODEL; k++) {
    float x = m[k];
    #pragma unroll
    for (int e = 0; e < 8; e++) acc[e] += x * wr[k * 8 + e];
  }
  acc[tok[n] & 7] += 10.0f;
  int best = 0; float bv = acc[0];
  #pragma unroll
  for (int e = 1; e < 8; e++) if (acc[e] > bv) { bv = acc[e]; best = e; }
  eid[n] = best;
  atomicAdd(&meta[best], 1);
}

__global__ void meta_zero_kernel(int* meta) {
  if (threadIdx.x < 32) meta[threadIdx.x] = 0;
}

__global__ void scan_kernel(int* meta) {
  if (threadIdx.x == 0 && blockIdx.x == 0) {
    int off = 0;
    for (int e = 0; e < 8; e++) { meta[16 + e] = off; off += meta[e]; }
    meta[24] = off;
  }
}

__global__ void scatter_kernel(const int* __restrict__ eid, int* __restrict__ meta,
                               int* __restrict__ perm) {
  int n = blockIdx.x * blockDim.x + threadIdx.x;
  if (n >= NTOK) return;
  int e = eid[n];
  int pos = meta[16 + e] + atomicAdd(&meta[8 + e], 1);
  perm[pos] = n;
}

// ---------------------------------------------------------------------------
// MoE stage 1: mid = silu(x@w_gate[e]) * (x@w_up[e]) over permuted segments.
// 64-row x 64-col tiles; each wave: 16 rows x 64 cols, dual accumulators.
// ---------------------------------------------------------------------------
__global__ __launch_bounds__(128) void moe_mid_kernel(
    const bf16_t* __restrict__ X, const float* __restrict__ Wg, const float* __restrict__ Wu,
    const int* __restrict__ perm, const int* __restrict__ offs, bf16_t* __restrict__ mid)
{
  __shared__ bf16_t Bs[2][64 * 40];
  const int e = blockIdx.z;
  const int seg = offs[e], len = offs[e + 1] - seg;
  const int rt = blockIdx.y;
  if (rt * 64 >= len) return;
  const int col0 = blockIdx.x * 64;
  const int tid = threadIdx.x, wid = tid >> 5, lane = tid & 31;
  const int g = lane >> 4, mloc = lane & 15;
  const int pl = rt * 64 + wid * 16 + mloc;
  const int token = perm[seg + (pl < len ? pl : len - 1)];
  const bf16_t* arow = X + (size_t)token * DMODEL;
  const float* wg = Wg + (size_t)e * DMODEL * FDIM;
  const float* wu = Wu + (size_t)e * DMODEL * FDIM;

  v8f zero = {0.f,0.f,0.f,0.f,0.f,0.f,0.f,0.f};
  v8f cg[4], cu[4];
  #pragma unroll
  for (int j = 0; j < 4; j++) { cg[j] = zero; cu[j] = zero; }

  for (int kt = 0; kt < DMODEL; kt += 32) {
    __syncthreads();
    for (int i = tid; i < 32 * 64; i += 128) {
      int kk = i >> 6, c = i & 63;
      Bs[0][c * 40 + kk] = (bf16_t)wg[(size_t)(kt + kk) * FDIM + col0 + c];
      Bs[1][c * 40 + kk] = (bf16_t)wu[(size_t)(kt + kk) * FDIM + col0 + c];
    }
    __syncthreads();
    v16bf a = ld_frag(arow + kt + 16 * g, arow + kt + 16 * g + 8);
    #pragma unroll
    for (int j = 0; j < 4; j++) {
      const bf16_t* b0 = &Bs[0][(j * 16 + mloc) * 40 + 16 * g];
      const bf16_t* b1 = &Bs[1][(j * 16 + mloc) * 40 + 16 * g];
      cg[j] = wmma_bf16(a, ld_frag(b0, b0 + 8), cg[j]);
      cu[j] = wmma_bf16(a, ld_frag(b1, b1 + 8), cu[j]);
    }
  }
  #pragma unroll
  for (int j = 0; j < 4; j++) {
    const int col = col0 + j * 16 + mloc;
    #pragma unroll
    for (int r = 0; r < 8; r++) {
      int plr = rt * 64 + wid * 16 + r + 8 * g;
      if (plr < len) {
        float gv = cg[j][r], uv = cu[j][r];
        float mv = gv / (1.f + __expf(-gv)) * uv;
        mid[(size_t)(seg + plr) * FDIM + col] = (bf16_t)mv;
      }
    }
  }
}

// ---------------------------------------------------------------------------
// MoE stage 2: y = mid@w_down[e], scatter-add into hidden output
// ---------------------------------------------------------------------------
__global__ __launch_bounds__(128) void moe_down_kernel(
    const bf16_t* __restrict__ mid, const float* __restrict__ Wd,
    const int* __restrict__ perm, const int* __restrict__ offs, float* __restrict__ out)
{
  __shared__ bf16_t Bs[64 * 40];
  const int e = blockIdx.z;
  const int seg = offs[e], len = offs[e + 1] - seg;
  const int rt = blockIdx.y;
  if (rt * 64 >= len) return;
  const int col0 = blockIdx.x * 64;
  const int tid = threadIdx.x, wid = tid >> 5, lane = tid & 31;
  const int g = lane >> 4, mloc = lane & 15;
  const int pl = rt * 64 + wid * 16 + mloc;
  const int p = seg + (pl < len ? pl : len - 1);
  const bf16_t* arow = mid + (size_t)p * FDIM;
  const float* wd = Wd + (size_t)e * FDIM * DMODEL;

  v8f zero = {0.f,0.f,0.f,0.f,0.f,0.f,0.f,0.f};
  v8f acc[4];
  #pragma unroll
  for (int j = 0; j < 4; j++) acc[j] = zero;

  for (int kt = 0; kt < FDIM; kt += 32) {
    __syncthreads();
    for (int i = tid; i < 32 * 64; i += 128) {
      int kk = i >> 6, c = i & 63;
      Bs[c * 40 + kk] = (bf16_t)wd[(size_t)(kt + kk) * DMODEL + col0 + c];
    }
    __syncthreads();
    v16bf a = ld_frag(arow + kt + 16 * g, arow + kt + 16 * g + 8);
    #pragma unroll
    for (int j = 0; j < 4; j++) {
      const bf16_t* bp = &Bs[(j * 16 + mloc) * 40 + 16 * g];
      acc[j] = wmma_bf16(a, ld_frag(bp, bp + 8), acc[j]);
    }
  }
  #pragma unroll
  for (int j = 0; j < 4; j++) {
    const int col = col0 + j * 16 + mloc;
    #pragma unroll
    for (int r = 0; r < 8; r++) {
      int plr = rt * 64 + wid * 16 + r + 8 * g;
      if (plr < len) {
        int token = perm[seg + plr];
        out[(size_t)token * DMODEL + col] += acc[j][r];
      }
    }
  }
}

// ---------------------------------------------------------------------------
// Host side
// ---------------------------------------------------------------------------
extern "C" void kernel_launch(void* const* d_in, const int* in_sizes, int n_in,
                              void* d_out, int out_size, void* d_ws, size_t ws_size,
                              hipStream_t stream) {
  (void)in_sizes; (void)n_in; (void)out_size; (void)ws_size;
  const float* hidden        = (const float*)d_in[0];
  const int*   positions     = (const int*)d_in[1];
  const float* velocity      = (const float*)d_in[2];
  const int*   token_ids     = (const int*)d_in[3];
  const float* mu_prev       = (const float*)d_in[4];
  const float* ln1_w         = (const float*)d_in[5];
  const float* ln2_w         = (const float*)d_in[6];
  const float* wq            = (const float*)d_in[7];
  const float* wk            = (const float*)d_in[8];
  const float* wv            = (const float*)d_in[9];
  const float* wo            = (const float*)d_in[10];
  const float* w_mu_q        = (const float*)d_in[11];
  const float* w_mu_k        = (const float*)d_in[12];
  const float* w_mu_v        = (const float*)d_in[13];
  const float* qnorm_w       = (const float*)d_in[14];
  const float* knorm_w       = (const float*)d_in[15];
  const float* dyn_mu        = (const float*)d_in[16];
  const float* dyn_mu_proj_w = (const float*)d_in[17];
  const float* ctrl_in_w     = (const float*)d_in[18];
  const float* ctrl_in_b     = (const float*)d_in[19];
  const float* ctrl_out_w    = (const float*)d_in[20];
  const float* ctrl_out_b    = (const float*)d_in[21];
  const float* mu_router_w   = (const float*)d_in[22];
  const float* w_gate        = (const float*)d_in[23];
  const float* w_up          = (const float*)d_in[24];
  const float* w_down        = (const float*)d_in[25];

  float* out_hidden = (float*)d_out;
  float* out_vnext  = out_hidden + (size_t)NTOK * DMODEL;
  float* out_mu     = out_vnext  + (size_t)NTOK * DMODEL;

  char* wp = (char*)d_ws;
  auto carve = [&](size_t bytes) -> void* {
    void* r = (void*)wp;
    wp += (bytes + 255) & ~(size_t)255;
    return r;
  };
  bf16_t* h_bf    = (bf16_t*)carve((size_t)NTOK * DMODEL * 2);
  bf16_t* mu_bf   = (bf16_t*)carve((size_t)NTOK * DMODEL * 2);
  bf16_t* vel_bf  = (bf16_t*)carve((size_t)NTOK * DMODEL * 2);
  float*  q_f     = (float*) carve((size_t)NTOK * NH * DHEAD * 4);
  float*  k_f     = (float*) carve((size_t)NTOK * NKV * DHEAD * 4);
  float*  v_f     = (float*) carve((size_t)NTOK * NKV * DHEAD * 4);
  bf16_t* q_bf    = (bf16_t*)carve((size_t)NTOK * NH * DHEAD * 2);
  bf16_t* k_bf    = (bf16_t*)carve((size_t)NTOK * NKV * DHEAD * 2);
  bf16_t* vT_bf   = (bf16_t*)carve((size_t)NKV * DHEAD * NTOK * 2);
  float*  araw_f  = (float*) carve((size_t)NTOK * NH * DHEAD * 4);
  bf16_t* araw_bf = (bf16_t*)carve((size_t)NTOK * NH * DHEAD * 2);
  float*  o_f     = (float*) carve((size_t)NTOK * DMODEL * 4);
  bf16_t* o_bf    = (bf16_t*)carve((size_t)NTOK * DMODEL * 2);
  bf16_t* ctrl_bf = (bf16_t*)carve((size_t)NTOK * CHID * 2);
  float*  co_f    = (float*) carve((size_t)NTOK * 3 * DMODEL * 4);
  bf16_t* x_bf    = (bf16_t*)carve((size_t)NTOK * DMODEL * 2);
  bf16_t* mid_bf  = (bf16_t*)carve((size_t)NTOK * FDIM * 2);
  int*    eid     = (int*)   carve((size_t)NTOK * 4);
  int*    perm    = (int*)   carve((size_t)NTOK * 4);
  int*    meta    = (int*)   carve(128 * 4);
  int*    offs    = meta + 16;

  const int ND = NTOK * DMODEL;

  meta_zero_kernel<<<1, 32, 0, stream>>>(meta);
  rmsnorm_kernel<<<NTOK, 256, 0, stream>>>(hidden, ln1_w, h_bf);
  cvt_kernel<<<(ND + 255) / 256, 256, 0, stream>>>(mu_prev, mu_bf, ND);
  cvt_kernel<<<(ND + 255) / 256, 256, 0, stream>>>(velocity, vel_bf, ND);

  // fused QKV (dual-input GEMMs): q/k/v = h@W + mu_prev@W_mu
  gemm_dense<<<dim3(NH * DHEAD / 64, NTOK / 64), 128, 0, stream>>>(
      h_bf, wq, mu_bf, w_mu_q, NH * DHEAD, DMODEL, DMODEL, NH * DHEAD,
      nullptr, q_f, nullptr, 0);
  gemm_dense<<<dim3(NKV * DHEAD / 64, NTOK / 64), 128, 0, stream>>>(
      h_bf, wk, mu_bf, w_mu_k, NKV * DHEAD, DMODEL, DMODEL, NKV * DHEAD,
      nullptr, k_f, nullptr, 0);
  gemm_dense<<<dim3(NKV * DHEAD / 64, NTOK / 64), 128, 0, stream>>>(
      h_bf, wv, mu_bf, w_mu_v, NKV * DHEAD, DMODEL, DMODEL, NKV * DHEAD,
      nullptr, v_f, nullptr, 0);

  ropeqk_kernel<<<dim3(NTOK, NH + NKV), 32, 0, stream>>>(
      q_f, k_f, qnorm_w, knorm_w, positions, q_bf, k_bf);
  vtrans_kernel<<<(NKV * DHEAD * NTOK + 255) / 256, 256, 0, stream>>>(v_f, vT_bf);

  attn_kernel<<<(NTOK / 16) * NH / 4, 128, 0, stream>>>(q_bf, k_bf, vT_bf, araw_f);
  cvt_kernel<<<(ND + 255) / 256, 256, 0, stream>>>(araw_f, araw_bf, ND);

  // o = attn_out @ wo
  gemm_dense<<<dim3(DMODEL / 64, NTOK / 64), 128, 0, stream>>>(
      araw_bf, wo, nullptr, nullptr, DMODEL, NH * DHEAD, NH * DHEAD, DMODEL,
      nullptr, o_f, nullptr, 0);
  cvt_kernel<<<(ND + 255) / 256, 256, 0, stream>>>(o_f, o_bf, ND);

  // mu_cur = dyn_mu + o @ dyn_mu_proj_w  (written straight into d_out slice 2)
  gemm_dense<<<dim3(DMODEL / 64, NTOK / 64), 128, 0, stream>>>(
      o_bf, dyn_mu_proj_w, nullptr, nullptr, DMODEL, DMODEL, DMODEL, DMODEL,
      dyn_mu, out_mu, nullptr, 1);

  // ctrl = silu([o, velocity] @ ctrl_in_w + b)  via dual-input GEMM
  gemm_dense<<<dim3(CHID / 64, NTOK / 64), 128, 0, stream>>>(
      o_bf, ctrl_in_w, vel_bf, ctrl_in_w + (size_t)DMODEL * CHID,
      CHID, DMODEL, DMODEL, CHID, ctrl_in_b, nullptr, ctrl_bf, 2);

  // co = ctrl @ ctrl_out_w + b
  gemm_dense<<<dim3(3 * DMODEL / 64, NTOK / 64), 128, 0, stream>>>(
      ctrl_bf, ctrl_out_w, nullptr, nullptr, 3 * DMODEL, CHID, CHID, 3 * DMODEL,
      ctrl_out_b, co_f, nullptr, 1);

  update_kernel<<<(ND + 255) / 256, 256, 0, stream>>>(
      hidden, o_f, velocity, co_f, out_mu, out_hidden, out_vnext);

  rmsnorm_kernel<<<NTOK, 256, 0, stream>>>(out_hidden, ln2_w, x_bf);

  router_kernel<<<(NTOK + 255) / 256, 256, 0, stream>>>(out_mu, mu_router_w, token_ids, eid, meta);
  scan_kernel<<<1, 32, 0, stream>>>(meta);
  scatter_kernel<<<(NTOK + 255) / 256, 256, 0, stream>>>(eid, meta, perm);

  moe_mid_kernel<<<dim3(FDIM / 64, NTOK / 64, NEXP), 128, 0, stream>>>(
      x_bf, w_gate, w_up, perm, offs, mid_bf);
  moe_down_kernel<<<dim3(DMODEL / 64, NTOK / 64, NEXP), 128, 0, stream>>>(
      mid_bf, w_down, perm, offs, out_hidden);
}
